// NonlocalAttentionFVC_19988777795685
// MI455X (gfx1250) — matrix operand order repaired and checked by reading, not compiled, not run on hardware
//
#include <hip/hip_runtime.h>
#include <hip/hip_bf16.h>

// ---------------------------------------------------------------------------
// Nonlocal attention (FVC), MI455X / gfx1250, wave32, bf16 WMMA.
//   B=8, C=128, H=W=64, N=4096.
//   softmax over axis 1 (ref positions) == per-COLUMN softmax of S[n,m].
//   => z[n,:] = sum_m exp(S[n,m]-M[m]) / D[m] * x[m,:], with column stats
//      (M,D) computed once in pass 1. Flash-style: S is never materialized
//      (537MB avoided); x/y are 1MB/batch bf16 and stay resident in L2.
// Scratch layout in d_ws (needs ~33 MB):
//   x_nc  [B][N][C] bf16   (theta(ref), pos-major)
//   x_cn  [B][C][N] bf16   (theta(ref), chan-major, B-operand of z-WMMA)
//   y_mc  [B][N][C] bf16   (phi(cur),  pos-major,  A-operand of S^T-WMMA)
//   zT    [B][C][N] bf16
//   stats [B][N] float2 {colmax, 1/colsum}
// ---------------------------------------------------------------------------

#define BB 8
#define CC 128
#define NN 4096

typedef __attribute__((ext_vector_type(16))) __bf16 v16bf;
typedef __attribute__((ext_vector_type(8)))  __bf16 v8bf;
typedef __attribute__((ext_vector_type(8)))  float  v8f;

__device__ __forceinline__ v16bf cat16(v8bf lo, v8bf hi) {
    return __builtin_shufflevector(lo, hi, 0,1,2,3,4,5,6,7,8,9,10,11,12,13,14,15);
}
__device__ __forceinline__ v8f wmma_bf16(v16bf a, v16bf b, v8f c) {
    // v_wmma_f32_16x16x32_bf16  D = A*B + C
    return __builtin_amdgcn_wmma_f32_16x16x32_bf16(false, a, false, b, (short)0, c, false, false);
}
__device__ __forceinline__ v8bf ldg8(const __bf16* p) {
    return *reinterpret_cast<const v8bf*>(p);
}
__device__ __forceinline__ v8f vzero8() {
    v8f z = {0.f,0.f,0.f,0.f,0.f,0.f,0.f,0.f};
    return z;
}

// ---------------------------------------------------------------------------
// Kernel 1: 1x1 conv projection. out[n,c_out] = sum_cin w[c_out,cin]*src[cin,n]+b
// Block: 256 thr, one (batch, 32-pos tile). Wave w handles couts [16w,16w+16),
// lane = position j. Weight indices are wave-uniform (readfirstlane) -> s_loads.
// ---------------------------------------------------------------------------
__global__ void proj_kernel(const float* __restrict__ src,   // [B][C][N]
                            const float* __restrict__ w,     // [C][C]
                            const float* __restrict__ bias,  // [C]
                            __bf16* __restrict__ out_nc,     // [B][N][C]
                            __bf16* __restrict__ out_cn,     // [B][C][N] or null
                            int write_cn)
{
    __shared__ float tile[CC * 32];                 // [cin][j], 16 KB
    const int b  = blockIdx.y;
    const int n0 = blockIdx.x * 32;
    const float* srcb = src + (size_t)b * CC * NN;

    for (int k = threadIdx.x; k < CC * 32; k += blockDim.x) {
        int cin = k >> 5, j = k & 31;
        tile[k] = srcb[(size_t)cin * NN + n0 + j];
    }
    __syncthreads();

    const int lane = threadIdx.x & 31;
    const int wid  = __builtin_amdgcn_readfirstlane((int)(threadIdx.x >> 5)); // 0..7
    const int coutBase = wid * 16;

    float acc[16];
#pragma unroll
    for (int u = 0; u < 16; ++u) acc[u] = bias[coutBase + u];

    for (int cin = 0; cin < CC; ++cin) {
        float r = tile[cin * 32 + lane];
#pragma unroll
        for (int u = 0; u < 16; ++u)
            acc[u] = fmaf(w[(coutBase + u) * CC + cin], r, acc[u]);
    }

    // pos-major write: 16 contiguous bf16 = two b128 stores
    v8bf p0, p1;
#pragma unroll
    for (int u = 0; u < 8; ++u) { p0[u] = (__bf16)acc[u]; p1[u] = (__bf16)acc[8 + u]; }
    __bf16* dst = out_nc + (size_t)b * NN * CC + (size_t)(n0 + lane) * CC + coutBase;
    *reinterpret_cast<v8bf*>(dst)     = p0;
    *reinterpret_cast<v8bf*>(dst + 8) = p1;

    if (write_cn) {
        __bf16* dct = out_cn + (size_t)b * NN * CC;
#pragma unroll
        for (int u = 0; u < 16; ++u)
            dct[(size_t)(coutBase + u) * NN + n0 + lane] = (__bf16)acc[u];
    }
}

// ---------------------------------------------------------------------------
// Kernel 2: column softmax stats. One wave per (batch, 16-wide m strip).
// S^T[m,n] tiles via WMMA (A = y_mc rows m, B = x_nc rows n).
// TWO passes over n (S recompute is L2-resident and rides the matrix pipe):
//   pass A: column max only      (dual-issue v_max, zero transcendentals)
//   pass B: sum of exp(v - M)    (exactly 1 v_exp per element, no dep chain)
// This halves exp count vs online softmax and removes the serial rescale.
// ---------------------------------------------------------------------------
__global__ void colstats_kernel(const __bf16* __restrict__ x_nc,
                                const __bf16* __restrict__ y_mc,
                                float2* __restrict__ stats)
{
    const int lane  = threadIdx.x & 31;
    const int gwave = (int)(blockIdx.x * (blockDim.x >> 5)) + (int)(threadIdx.x >> 5);
    const int b  = gwave >> 8;            // 256 strips / batch
    const int m0 = (gwave & 255) * 16;
    const int l16  = lane & 15;
    const int half = lane >> 4;

    const __bf16* xb = x_nc + (size_t)b * NN * CC;
    const __bf16* yb = y_mc + (size_t)b * NN * CC;

    // Preload A operand (y rows m0..m0+15, K = all 128 channels, 4 k-steps)
    v16bf A[4];
    {
        const __bf16* row = yb + (size_t)(m0 + l16) * CC;
#pragma unroll
        for (int kk = 0; kk < 4; ++kk) {
            const __bf16* p = row + 32 * kk + (half ? 16 : 0);
            A[kk] = cat16(ldg8(p), ldg8(p + 8));
        }
    }

    // ---- pass A: column max ----
    float rmax[8];
#pragma unroll
    for (int r = 0; r < 8; ++r) rmax[r] = -1.0e30f;

    for (int nc = 0; nc < NN; nc += 16) {
        v8f acc = vzero8();
        const __bf16* brow = xb + (size_t)(nc + l16) * CC;
#pragma unroll
        for (int kk = 0; kk < 4; ++kk) {
            const __bf16* p = brow + 32 * kk + (half ? 16 : 0);
            v16bf Bv = cat16(ldg8(p), ldg8(p + 8));
            acc = wmma_bf16(A[kk], Bv, acc);
        }
#pragma unroll
        for (int r = 0; r < 8; ++r) rmax[r] = fmaxf(rmax[r], acc[r]);
    }
    // merge max across the 16 lanes of each half-wave -> final column max M
#pragma unroll
    for (int mask = 1; mask < 16; mask <<= 1) {
#pragma unroll
        for (int r = 0; r < 8; ++r)
            rmax[r] = fmaxf(rmax[r], __shfl_xor(rmax[r], mask, 32));
    }

    // ---- pass B: sum of exp(v - M) ----
    float rsum[8];
#pragma unroll
    for (int r = 0; r < 8; ++r) rsum[r] = 0.f;

    for (int nc = 0; nc < NN; nc += 16) {
        v8f acc = vzero8();
        const __bf16* brow = xb + (size_t)(nc + l16) * CC;
#pragma unroll
        for (int kk = 0; kk < 4; ++kk) {
            const __bf16* p = brow + 32 * kk + (half ? 16 : 0);
            v16bf Bv = cat16(ldg8(p), ldg8(p + 8));
            acc = wmma_bf16(A[kk], Bv, acc);
        }
#pragma unroll
        for (int r = 0; r < 8; ++r) rsum[r] += __expf(acc[r] - rmax[r]);
    }
    // merge sums (plain adds)
#pragma unroll
    for (int mask = 1; mask < 16; mask <<= 1) {
#pragma unroll
        for (int r = 0; r < 8; ++r)
            rsum[r] += __shfl_xor(rsum[r], mask, 32);
    }

    if (l16 == 0) {
        float2* sp = stats + (size_t)b * NN + m0 + half * 8;
#pragma unroll
        for (int r = 0; r < 8; ++r)
            sp[r] = make_float2(rmax[r], 1.0f / rsum[r]);
    }
}

// ---------------------------------------------------------------------------
// Kernel 3: fused pass 2. One wave per (batch, 16-row n strip).
// Per 32-wide m chunk: recompute S^T (8 WMMA), exp*rcpD, pack P to LDS
// (2 packed ds_store_b128/lane), reload as A operand (2 ds_load_b128/lane),
// then 8 z-WMMAs into a 64-VGPR [16n x 128c] accumulator. zT written bf16.
// global_prefetch of next chunk's y / xT rows overlaps L2 fetch with WMMAs.
// ---------------------------------------------------------------------------
__global__ void attn_kernel(const __bf16* __restrict__ x_nc,
                            const __bf16* __restrict__ x_cn,
                            const __bf16* __restrict__ y_mc,
                            const float2* __restrict__ stats,
                            __bf16* __restrict__ zT)
{
    __shared__ __align__(16) __bf16 Pbuf[4][16 * 32];   // per-wave [n][m] tile

    const int wslot = (int)(threadIdx.x >> 5);
    const int lane  = threadIdx.x & 31;
    const int gwave = (int)(blockIdx.x * (blockDim.x >> 5)) + wslot;
    const int b  = gwave >> 8;
    const int n0 = (gwave & 255) * 16;
    const int l16  = lane & 15;
    const int half = lane >> 4;

    const __bf16*  xb  = x_nc + (size_t)b * NN * CC;
    const __bf16*  xTb = x_cn + (size_t)b * NN * CC;
    const __bf16*  yb  = y_mc + (size_t)b * NN * CC;
    const float2*  stb = stats + (size_t)b * NN;
    __bf16* Prow = &Pbuf[wslot][0];

    // Preload B operand of S^T: x rows n0..n0+15, all 4 k-steps (fixed all kernel)
    v16bf Bx[4];
    {
        const __bf16* row = xb + (size_t)(n0 + l16) * CC;
#pragma unroll
        for (int kk = 0; kk < 4; ++kk) {
            const __bf16* p = row + 32 * kk + (half ? 16 : 0);
            Bx[kk] = cat16(ldg8(p), ldg8(p + 8));
        }
    }

    v8f zacc[8];
#pragma unroll
    for (int ct = 0; ct < 8; ++ct) zacc[ct] = vzero8();

    for (int mc = 0; mc < NN; mc += 32) {
        // prefetch next chunk's operand rows into cache (global_prefetch_b8)
        if (mc + 32 < NN) {
            __builtin_prefetch(yb + (size_t)(mc + 32 + lane) * CC, 0, 1);
            __builtin_prefetch(xTb + (size_t)(lane * 4 + l16 / 4) * NN + mc + 32, 0, 1);
        }

        // ---- compute P tile [16n x 32m] and stage into LDS ----
#pragma unroll
        for (int sub = 0; sub < 2; ++sub) {
            const int ms = mc + sub * 16;
            v8f sacc = vzero8();
            const __bf16* arow = yb + (size_t)(ms + l16) * CC;
#pragma unroll
            for (int kk = 0; kk < 4; ++kk) {
                const __bf16* p = arow + 32 * kk + (half ? 16 : 0);
                v16bf Av = cat16(ldg8(p), ldg8(p + 8));
                sacc = wmma_bf16(Av, Bx[kk], sacc);
            }
            // column stats for m = ms + half*8 + r  (8 consecutive float2)
            float2 st[8];
            {
                const float4* sp = reinterpret_cast<const float4*>(stb + ms + half * 8);
                float4* dst4 = reinterpret_cast<float4*>(st);
#pragma unroll
                for (int q = 0; q < 4; ++q) dst4[q] = sp[q];
            }
            v8bf pv;
#pragma unroll
            for (int r = 0; r < 8; ++r)
                pv[r] = (__bf16)(__expf(sacc[r] - st[r].x) * st[r].y);
            // element (n=l16, m=sub*16+half*8+r): consecutive r -> one b128 store
            *reinterpret_cast<v8bf*>(Prow + l16 * 32 + sub * 16 + half * 8) = pv;
        }
        __syncthreads();

        // ---- reload P as A operand ([M=n][K=m] layout) ----
        v16bf Ap;
        {
            const __bf16* p = Prow + l16 * 32 + (half ? 16 : 0);
            Ap = cat16(*reinterpret_cast<const v8bf*>(p),
                       *reinterpret_cast<const v8bf*>(p + 8));
        }

        // ---- z += P * x   (B operand = xT rows c, K = m chunk) ----
#pragma unroll
        for (int ct = 0; ct < 8; ++ct) {
            const __bf16* p = xTb + (size_t)(ct * 16 + l16) * NN + mc + (half ? 16 : 0);
            v16bf Bz = cat16(ldg8(p), ldg8(p + 8));
            zacc[ct] = wmma_bf16(Ap, Bz, zacc[ct]);
        }
        __syncthreads();
    }

    // write zT[c][n] bf16: lane holds z[n=half*8+r, c=ct*16+l16] -> packed b128
    __bf16* zb = zT + (size_t)b * NN * CC;
#pragma unroll
    for (int ct = 0; ct < 8; ++ct) {
        v8bf ov;
#pragma unroll
        for (int r = 0; r < 8; ++r) ov[r] = (__bf16)zacc[ct][r];
        *reinterpret_cast<v8bf*>(zb + (size_t)(ct * 16 + l16) * NN + n0 + half * 8) = ov;
    }
}

// ---------------------------------------------------------------------------
// Kernel 4: output 1x1 conv. out[b,cout,n] = Wb[cout] + sum_c Ww[cout,c]*zT[c,n]
// cout from blockIdx -> weight loads are scalar; zT loads coalesced over n.
// ---------------------------------------------------------------------------
__global__ void outproj_kernel(const __bf16* __restrict__ zT,
                               const float* __restrict__ w,
                               const float* __restrict__ bias,
                               float* __restrict__ out)
{
    const int b    = blockIdx.z;
    const int cout = blockIdx.y;
    const int n    = blockIdx.x * 256 + threadIdx.x;
    const __bf16* zb = zT + (size_t)b * NN * CC;
    const float*  wr = w + (size_t)cout * CC;

    float acc = bias[cout];
#pragma unroll 8
    for (int c = 0; c < CC; ++c)
        acc = fmaf((float)zb[(size_t)c * NN + n], wr[c], acc);

    out[((size_t)b * CC + cout) * NN + n] = acc;
}

// ---------------------------------------------------------------------------
extern "C" void kernel_launch(void* const* d_in, const int* in_sizes, int n_in,
                              void* d_out, int out_size, void* d_ws, size_t ws_size,
                              hipStream_t stream)
{
    const float* cur     = (const float*)d_in[0];
    const float* ref     = (const float*)d_in[1];
    const float* theta_w = (const float*)d_in[2];
    const float* theta_b = (const float*)d_in[3];
    const float* phi_w   = (const float*)d_in[4];
    const float* phi_b   = (const float*)d_in[5];
    const float* W_w     = (const float*)d_in[6];
    const float* W_b     = (const float*)d_in[7];

    char* ws = (char*)d_ws;
    const size_t SZ = (size_t)BB * NN * CC * sizeof(__bf16);   // 8 MB per array
    __bf16* x_nc = (__bf16*)(ws);
    __bf16* x_cn = (__bf16*)(ws + SZ);
    __bf16* y_mc = (__bf16*)(ws + 2 * SZ);
    __bf16* zTs  = (__bf16*)(ws + 3 * SZ);
    float2* stats = (float2*)(ws + 4 * SZ);

    dim3 gProj(NN / 32, BB);
    proj_kernel<<<gProj, 256, 0, stream>>>(ref, theta_w, theta_b, x_nc, x_cn, 1);
    proj_kernel<<<gProj, 256, 0, stream>>>(cur, phi_w, phi_b, y_mc, nullptr, 0);

    // 2048 waves total, 4 waves/block
    colstats_kernel<<<512, 128, 0, stream>>>(x_nc, y_mc, stats);
    attn_kernel<<<512, 128, 0, stream>>>(x_nc, x_cn, y_mc, stats, zTs);

    outproj_kernel<<<dim3(NN / 256, CC, BB), 256, 0, stream>>>(zTs, W_w, W_b, (float*)d_out);
}
